// BaselineClassifier_14774687498496
// MI455X (gfx1250) — matrix operand order repaired and checked by reading, not compiled, not run on hardware
//
#include <hip/hip_runtime.h>
#include <hip/hip_bf16.h>
#include <math.h>

// ---------------- static problem config (matches reference) ----------------
#define BATCH   128
#define NS      8
#define NSAMP   (BATCH * NS)        // 1024 rows
#define SEQ     337
#define FEAT    3
#define KSZ     5
#define F1      128
#define DU      256
#define ALPHA   0.3f

typedef unsigned short ushort_t;
typedef unsigned int   uint_t;

typedef __attribute__((ext_vector_type(16))) __bf16 v16bf;
typedef __attribute__((ext_vector_type(8)))  float  v8f;

union FragBF {            // 16 bf16 = 32 bytes = 2 x uint4
    v16bf v;
    uint4 q[2];
};

__device__ __forceinline__ ushort_t f2bf(float f) {
    union { float f; uint_t u; } c; c.f = f;
    uint_t r = c.u + 0x7FFFu + ((c.u >> 16) & 1u);   // round-to-nearest-even
    return (ushort_t)(r >> 16);
}
__device__ __forceinline__ float bf2f(ushort_t s) {
    union { uint_t u; float f; } c; c.u = ((uint_t)s) << 16;
    return c.f;
}
__device__ __forceinline__ float lrelu(float x) { return x >= 0.f ? x : ALPHA * x; }

// ---------------------------------------------------------------------------
// Weight prep: f32 [R, C] (R = 5*Cin or K) -> bf16 transposed [C, R]
// Makes every WMMA B-fragment two contiguous global_load_b128 per lane.
// ---------------------------------------------------------------------------
__global__ void wprep_kernel(const float* __restrict__ w, ushort_t* __restrict__ o,
                             int R, int C) {
    int idx = blockIdx.x * 256 + threadIdx.x;
    if (idx >= R * C) return;
    int c = idx % C, r = idx / C;                 // coalesced read
    o[(size_t)c * R + r] = f2bf(w[idx]);
}

// ---------------------------------------------------------------------------
// Conv1 (Cin=3, tiny K-dim): direct VALU kernel; fuses graph_reshape
// (identity gather) + bias + LeakyReLU, emits bf16 activations.
// ---------------------------------------------------------------------------
__global__ void conv1_kernel(const float* __restrict__ feats,
                             const float* __restrict__ W1,  // [5,3,128]
                             const float* __restrict__ b1,
                             ushort_t* __restrict__ out) {  // [1024,337,128] bf16
    int l = blockIdx.x, n = blockIdx.y, co = threadIdx.x;   // co in [0,128)
    int b = n >> 3, ns = n & 7;
    float acc = b1[co];
    #pragma unroll
    for (int k = 0; k < KSZ; ++k) {
        int t = l + k - 2;
        if (t < 0 || t >= SEQ) continue;
        const float* xr = feats + ((size_t)(b * SEQ + t) * NS + ns) * FEAT;
        #pragma unroll
        for (int ci = 0; ci < FEAT; ++ci)
            acc = fmaf(xr[ci], W1[(k * FEAT + ci) * F1 + co], acc);
    }
    out[((size_t)n * SEQ + l) * F1 + co] = f2bf(lrelu(acc));
}

// ---------------------------------------------------------------------------
// Generic conv1d (SAME, K=5) as implicit GEMM with v_wmma_f32_16x16x32_bf16.
// Templated on CIN (static shifts, unrollable K-loop) and NACC (cout subtiles
// per wave, A-fragment reuse). Block = 256 thr = 8 waves; block covers
// 16 positions x (8*16*NACC) couts of one sample. Input tile
// [pos0-2 .. pos0+17] x CIN staged once into LDS (b128 copies), shared by all
// 8 waves. Wave tile = 16 pos x 16*NACC couts; K-loop = 5 taps x CIN/32.
//   A (16x32 bf16, LDS):       lane = M(pos), half -> K+16;  2 x ds_load_b128
//   B (32x16 bf16, weights^T): lane = N(cout), half -> K+16; 2 x global b128
// Bias folded into acc init; LeakyReLU fused in epilogue; bf16 output.
// ---------------------------------------------------------------------------
template <int CIN, int NACC>
__global__ void conv_wmma_kernel(const ushort_t* __restrict__ xin,  // [N,L,CIN] bf16
                                 const ushort_t* __restrict__ wt,   // [Cout,5*CIN] bf16
                                 const float*    __restrict__ bias, // [Cout]
                                 ushort_t*       __restrict__ out,  // [N,L,Cout] bf16
                                 int L, int Cout) {
    __shared__ ushort_t xs[20 * CIN];
    const int n    = blockIdx.x;
    const int pt   = blockIdx.y;                // 16-position tile
    const int tid  = threadIdx.x;
    const int wave = tid >> 5, lane = tid & 31;
    const int m = lane & 15, h = lane >> 4;
    const int co0 = (blockIdx.z * 8 + wave) * (16 * NACC);
    const int pos0 = pt * 16;

    // cooperative staged load: 20 rows x CIN bf16, zero-padded outside [0,L)
    constexpr int CW4 = CIN / 8;                // uint4 per row (power of two)
    const uint4* xin128 = (const uint4*)(xin + (size_t)n * L * CIN);
    uint4* xs128 = (uint4*)xs;
    #pragma unroll
    for (int d = tid; d < 20 * CW4; d += 256) {
        int row = d / CW4;                      // constexpr pow2 -> shift
        int cp  = d - row * CW4;
        int p   = pos0 - 2 + row;
        uint4 v = {0u, 0u, 0u, 0u};
        if (p >= 0 && p < L) v = xin128[(size_t)p * CW4 + cp];
        xs128[d] = v;
    }
    __syncthreads();

    v8f acc[NACC];
    #pragma unroll
    for (int j = 0; j < NACC; ++j) {
        const float bv = bias[co0 + 16 * j + m];
        #pragma unroll
        for (int i = 0; i < 8; ++i) acc[j][i] = bv;
    }

    const ushort_t* wrow[NACC];
    #pragma unroll
    for (int j = 0; j < NACC; ++j)
        wrow[j] = wt + (size_t)(co0 + 16 * j + m) * (KSZ * CIN);

    #pragma unroll
    for (int k = 0; k < KSZ; ++k) {
        const ushort_t* xrow = xs + (m + k) * CIN;   // output pos m, tap k
        for (int ci0 = 0; ci0 < CIN; ci0 += 32) {
            FragBF a;
            a.q[0] = *(const uint4*)(xrow + ci0 + 16 * h);       // K 0..7  (+16h)
            a.q[1] = *(const uint4*)(xrow + ci0 + 16 * h + 8);   // K 8..15 (+16h)
            #pragma unroll
            for (int j = 0; j < NACC; ++j) {
                FragBF b;
                const ushort_t* wp = wrow[j] + k * CIN + ci0 + 16 * h;
                b.q[0] = *(const uint4*)(wp);                    // K 0..7  (+16h)
                b.q[1] = *(const uint4*)(wp + 8);                // K 8..15 (+16h)
                acc[j] = __builtin_amdgcn_wmma_f32_16x16x32_bf16(
                    false, a.v, false, b.v, (short)0, acc[j], false, false);
            }
        }
    }

    // C layout: lanes0-15 VGPRv -> M=v; lanes16-31 -> M=8+v; N = lane&15
    #pragma unroll
    for (int j = 0; j < NACC; ++j) {
        ushort_t* orow = out + ((size_t)n * L + pos0) * Cout + co0 + 16 * j + m;
        #pragma unroll
        for (int v = 0; v < 8; ++v) {
            int p = v + 8 * h;
            if (pos0 + p < L)
                orow[(size_t)p * Cout] = f2bf(lrelu(acc[j][v]));
        }
    }
}

// ---------------------------------------------------------------------------
// MaxPool1D(2), stride 2, VALID — bf16, two channels per thread (dword).
// ---------------------------------------------------------------------------
__global__ void maxpool_kernel(const ushort_t* __restrict__ in, ushort_t* __restrict__ out,
                               int N, int Lin, int Lout, int C) {
    size_t idx = (size_t)blockIdx.x * 256 + threadIdx.x;
    int cw = C >> 1;
    size_t total = (size_t)N * Lout * cw;
    if (idx >= total) return;
    int cp = (int)(idx % cw);
    size_t r = idx / cw;
    int l = (int)(r % Lout);
    int n = (int)(r / Lout);
    const uint_t* a = (const uint_t*)(in + ((size_t)n * Lin + 2 * l) * C) + cp;
    uint_t x0 = a[0], x1 = a[cw];
    float a0 = bf2f((ushort_t)(x0 & 0xFFFF)), a1 = bf2f((ushort_t)(x0 >> 16));
    float b0 = bf2f((ushort_t)(x1 & 0xFFFF)), b1 = bf2f((ushort_t)(x1 >> 16));
    uint_t lo = (a0 >= b0) ? (x0 & 0xFFFF) : (x1 & 0xFFFF);
    uint_t hi = (a1 >= b1) ? (x0 >> 16)    : (x1 >> 16);
    ((uint_t*)out)[idx] = lo | (hi << 16);
}

// ---------------------------------------------------------------------------
// GlobalAveragePooling1D over L: [N,L,C] bf16 -> [N,C] bf16
// ---------------------------------------------------------------------------
__global__ void gap_kernel(const ushort_t* __restrict__ in, ushort_t* __restrict__ out,
                           int N, int L, int C) {
    size_t idx = (size_t)blockIdx.x * 256 + threadIdx.x;
    if (idx >= (size_t)N * C) return;
    int c = (int)(idx % C), n = (int)(idx / C);
    const ushort_t* p = in + (size_t)n * L * C + c;
    float s = 0.f;
    for (int l = 0; l < L; ++l) s += bf2f(p[(size_t)l * C]);
    out[idx] = f2bf(s / (float)L);
}

// ---------------------------------------------------------------------------
// Dense GEMM via WMMA: x [M,K] bf16, wt [N,K] bf16 (pre-transposed),
// bias f32, LeakyReLU, out [M,N] bf16. Wave = 16 x 32 tile (2 accumulators,
// A-fragment reuse); block = 8 waves covers 16 x 256. K fully static.
// ---------------------------------------------------------------------------
template <int K>
__global__ void dense_wmma_kernel(const ushort_t* __restrict__ x,
                                  const ushort_t* __restrict__ wt,
                                  const float*    __restrict__ bias,
                                  ushort_t*       __restrict__ out,
                                  int N) {
    const int mt   = blockIdx.x;
    const int wave = threadIdx.x >> 5, lane = threadIdx.x & 31;
    const int m = lane & 15, h = lane >> 4;
    const int n0 = wave * 32;                    // 8 waves x 32 couts = 256 = N

    v8f acc[2];
    #pragma unroll
    for (int j = 0; j < 2; ++j) {
        const float bv = bias[n0 + 16 * j + m];
        #pragma unroll
        for (int i = 0; i < 8; ++i) acc[j][i] = bv;
    }

    const ushort_t* xrow  = x  + (size_t)(mt * 16 + m) * K;
    const ushort_t* wrow0 = wt + (size_t)(n0 + m) * K;
    const ushort_t* wrow1 = wt + (size_t)(n0 + 16 + m) * K;
    for (int ci0 = 0; ci0 < K; ci0 += 32) {
        FragBF a, b0, b1;
        a.q[0]  = *(const uint4*)(xrow + ci0 + 16 * h);
        a.q[1]  = *(const uint4*)(xrow + ci0 + 16 * h + 8);
        b0.q[0] = *(const uint4*)(wrow0 + ci0 + 16 * h);
        b0.q[1] = *(const uint4*)(wrow0 + ci0 + 16 * h + 8);
        b1.q[0] = *(const uint4*)(wrow1 + ci0 + 16 * h);
        b1.q[1] = *(const uint4*)(wrow1 + ci0 + 16 * h + 8);
        acc[0] = __builtin_amdgcn_wmma_f32_16x16x32_bf16(
            false, a.v, false, b0.v, (short)0, acc[0], false, false);
        acc[1] = __builtin_amdgcn_wmma_f32_16x16x32_bf16(
            false, a.v, false, b1.v, (short)0, acc[1], false, false);
    }

    #pragma unroll
    for (int j = 0; j < 2; ++j) {
        ushort_t* orow = out + (size_t)(mt * 16) * N + n0 + 16 * j + m;
        #pragma unroll
        for (int v = 0; v < 8; ++v) {
            int p = v + 8 * h;
            orow[(size_t)p * N] = f2bf(lrelu(acc[j][v]));
        }
    }
}

// ---------------------------------------------------------------------------
// Head: out[n] = sigmoid(d2[n,:] . Wo + bo)
// ---------------------------------------------------------------------------
__global__ void head_kernel(const ushort_t* __restrict__ x, const float* __restrict__ Wo,
                            const float* __restrict__ bo, float* __restrict__ out,
                            int N, int K) {
    int n = blockIdx.x * 256 + threadIdx.x;
    if (n >= N) return;
    const ushort_t* xr = x + (size_t)n * K;
    float s = bo[0];
    for (int k = 0; k < K; ++k) s = fmaf(bf2f(xr[k]), Wo[k], s);
    out[n] = 1.f / (1.f + expf(-s));
}

// ---------------------------------------------------------------------------
// Launch: weight prep -> conv1 -> [conv/pool ping-pong in bf16] -> GAP ->
// dense x2 -> head. Workspace: 11MB bf16 weights + 2 x 88.4MB activation
// buffers (~188 MB total, all bf16 to halve HBM traffic and feed WMMA).
// ---------------------------------------------------------------------------
extern "C" void kernel_launch(void* const* d_in, const int* in_sizes, int n_in,
                              void* d_out, int out_size, void* d_ws, size_t ws_size,
                              hipStream_t stream) {
    const float* feats = (const float*)d_in[0];
    // d_in[1] sample_indices: identity partition layout -> folded into conv1 indexing
    const float* W1 = (const float*)d_in[2];  const float* b1 = (const float*)d_in[3];
    const float* W2 = (const float*)d_in[4];  const float* b2 = (const float*)d_in[5];
    const float* Wa0 = (const float*)d_in[6]; const float* ba0 = (const float*)d_in[7];
    const float* Wb0 = (const float*)d_in[8]; const float* bb0 = (const float*)d_in[9];
    const float* Wa1 = (const float*)d_in[10]; const float* ba1 = (const float*)d_in[11];
    const float* Wb1 = (const float*)d_in[12]; const float* bb1 = (const float*)d_in[13];
    const float* W4 = (const float*)d_in[14]; const float* b4 = (const float*)d_in[15];
    const float* Wd1 = (const float*)d_in[16]; const float* bd1 = (const float*)d_in[17];
    const float* Wd2 = (const float*)d_in[18]; const float* bd2 = (const float*)d_in[19];
    const float* Wo = (const float*)d_in[20]; const float* bo = (const float*)d_in[21];
    float* outp = (float*)d_out;

    ushort_t* ws = (ushort_t*)d_ws;
    // bf16 transposed weight segments (element offsets)
    ushort_t* w2  = ws + 0;        // 5*128*128   = 81920
    ushort_t* wa0 = ws + 81920;    // 5*128*256   = 163840
    ushort_t* wb0 = ws + 245760;   // 5*256*256   = 327680
    ushort_t* wa1 = ws + 573440;   // 5*256*512   = 655360
    ushort_t* wb1 = ws + 1228800;  // 5*512*512   = 1310720
    ushort_t* wc4 = ws + 2539520;  // 5*512*1024  = 2621440
    ushort_t* wd1 = ws + 5160960;  // 1024*256    = 262144
    ushort_t* wd2 = ws + 5423104;  // 256*256     = 65536
    // ping-pong activation buffers (44,171,264 bf16 = max of [1024,337,128])
    ushort_t* bufA = ws + 5488640;
    ushort_t* bufB = bufA + 44171264;

    auto wp = [&](const float* w, ushort_t* o, int R, int C) {
        int total = R * C;
        wprep_kernel<<<(total + 255) / 256, 256, 0, stream>>>(w, o, R, C);
    };
    wp(W2,  w2,  5 * 128, 128);
    wp(Wa0, wa0, 5 * 128, 256);
    wp(Wb0, wb0, 5 * 256, 256);
    wp(Wa1, wa1, 5 * 256, 512);
    wp(Wb1, wb1, 5 * 512, 512);
    wp(W4,  wc4, 5 * 512, 1024);
    wp(Wd1, wd1, 1024, 256);
    wp(Wd2, wd2, 256, 256);

    conv1_kernel<<<dim3(SEQ, NSAMP), F1, 0, stream>>>(feats, W1, b1, bufA);

    auto pool = [&](const ushort_t* xin, ushort_t* o, int Lin, int Lout, int C) {
        size_t total = (size_t)NSAMP * Lout * (C >> 1);
        maxpool_kernel<<<(unsigned)((total + 255) / 256), 256, 0, stream>>>(
            xin, o, NSAMP, Lin, Lout, C);
    };

    // conv2: Cin=128, Cout=128 -> NACC=1, block covers all 128 couts
    conv_wmma_kernel<128, 1><<<dim3(NSAMP, (337 + 15) / 16, 1), 256, 0, stream>>>(
        bufA, w2, b2, bufB, 337, 128);
    pool(bufB, bufA, 337, 168, 128);
    // a0: 128 -> 256 couts: NACC=2, one block-z covers 256
    conv_wmma_kernel<128, 2><<<dim3(NSAMP, (168 + 15) / 16, 1), 256, 0, stream>>>(
        bufA, wa0, ba0, bufB, 168, 256);
    // b0: 256 -> 256
    conv_wmma_kernel<256, 2><<<dim3(NSAMP, (168 + 15) / 16, 1), 256, 0, stream>>>(
        bufB, wb0, bb0, bufA, 168, 256);
    pool(bufA, bufB, 168, 84, 256);
    // a1: 256 -> 512: z = 512/256 = 2
    conv_wmma_kernel<256, 2><<<dim3(NSAMP, (84 + 15) / 16, 2), 256, 0, stream>>>(
        bufB, wa1, ba1, bufA, 84, 512);
    // b1: 512 -> 512
    conv_wmma_kernel<512, 2><<<dim3(NSAMP, (84 + 15) / 16, 2), 256, 0, stream>>>(
        bufA, wb1, bb1, bufB, 84, 512);
    pool(bufB, bufA, 84, 42, 512);
    // conv4: 512 -> 1024: z = 1024/256 = 4
    conv_wmma_kernel<512, 2><<<dim3(NSAMP, (42 + 15) / 16, 4), 256, 0, stream>>>(
        bufA, wc4, b4, bufB, 42, 1024);

    gap_kernel<<<(NSAMP * 1024 + 255) / 256, 256, 0, stream>>>(bufB, bufA, NSAMP, 42, 1024);

    dense_wmma_kernel<1024><<<dim3(NSAMP / 16, 1), 256, 0, stream>>>(bufA, wd1, bd1, bufB, DU);
    dense_wmma_kernel<256><<<dim3(NSAMP / 16, 1), 256, 0, stream>>>(bufB, wd2, bd2, bufA, DU);

    head_kernel<<<(NSAMP + 255) / 256, 256, 0, stream>>>(bufA, Wo, bo, outp, NSAMP, DU);
}
